// _DiffusionConv_26560077758759
// MI455X (gfx1250) — compile-verified, matrix operand order checked by reading, not run on hardware
//
#include <hip/hip_runtime.h>
#include <hip/hip_bf16.h>

// ---------------------------------------------------------------------------
// DiffusionConv (DCRNN-style graph conv) for MI455X / gfx1250.
// 4 dependent GEMMs  Y = alpha * (S @ X) - P  with S 4096x4096 (f32->f16,
// L2-resident at 33.5MB), X kept transposed in f16 so WMMA A and B fragments
// are both contiguous-K b128 loads. v_wmma_f32_16x16x32_f16, f32 accumulate.
// 32x32 wave tiles (4 WMMA / K-step from 2 A + 2 B frags) + depth-2 software
// pipeline (loads run two K-steps ahead of the WMMAs -> partial loadcnt
// waits only). Final projection [16384x352]x[352x64] also via WMMA + bias.
// ---------------------------------------------------------------------------

typedef __attribute__((ext_vector_type(16))) _Float16 v16h;
typedef __attribute__((ext_vector_type(8)))  _Float16 v8h;
typedef __attribute__((ext_vector_type(8)))  float    v8f;

#define NSER   4096
#define KDIM   4096
#define LDC    288      // padded column count (264 real) = 9 tiles of 32
#define NMATS  5
#define BN     16384    // B*N
#define FMPAD  352      // F*NUM_MATS = 330 padded to 11*32
#define MATSZ  ((size_t)NSER * LDC)   // floats per mat

static __device__ __forceinline__ v16h cat16(v8h lo, v8h hi) {
  return __builtin_shufflevector(lo, hi, 0,1,2,3,4,5,6,7,8,9,10,11,12,13,14,15);
}
// A/B 16-bit fragment: lane holds K = [kb..kb+7] and [kb+16..kb+23],
// contiguous in memory when the K dimension is innermost -> two b128 loads.
static __device__ __forceinline__ v16h load_frag(const _Float16* __restrict__ p) {
  v8h lo = *(const v8h*)p;
  v8h hi = *(const v8h*)(p + 16);
  return cat16(lo, hi);
}

#define WMMA(A, B, C) \
  __builtin_amdgcn_wmma_f32_16x16x32_f16(false, (A), false, (B), (short)0, (C), false, false)

// ---------------------------------------------------------------- f32 -> f16
__global__ __launch_bounds__(256) void k_f32_to_f16(const float* __restrict__ src,
                                                    _Float16* __restrict__ dst, int n8) {
  int i = blockIdx.x * blockDim.x + threadIdx.x;
  if (i >= n8) return;
  v8f v = *(const v8f*)(src + (size_t)i * 8);
  *(v8h*)(dst + (size_t)i * 8) = __builtin_convertvector(v, v8h);
}

// -------------------------------------------------- build x0 (f32 + f16^T)
// x0[n, f*4+b] = f<2 ? inputs[b, n*2+f] : state[b, n*64+(f-2)]
__global__ __launch_bounds__(256) void k_pack_x0(const float* __restrict__ inp,
                                                 const float* __restrict__ st,
                                                 float* __restrict__ mat0,
                                                 _Float16* __restrict__ xt0) {
  int idx = blockIdx.x * blockDim.x + threadIdx.x;   // exactly 288*4096 threads
  int n = idx & (NSER - 1);
  int c = idx >> 12;
  float v = 0.0f;
  if (c < 264) {
    int f = c >> 2, b = c & 3;
    v = (f < 2) ? inp[(size_t)b * 8192 + n * 2 + f]
                : st[(size_t)b * 262144 + n * 64 + (f - 2)];
  }
  mat0[(size_t)n * LDC + c] = v;
  xt0[(size_t)c * KDIM + n] = (_Float16)v;
}

// --------------------------------------------- main diffusion GEMM (WMMA)
// Per wave: 32x32 C tile (c00,c01,c10,c11); 4 WMMAs per K=32 step from
// 2 A + 2 B fragments; two fragment stages so loads lead WMMAs by 64 in K.
__global__ __launch_bounds__(256)
void k_diff_gemm(const _Float16* __restrict__ S, const _Float16* __restrict__ Xt,
                 const float* __restrict__ P, float alpha,
                 float* __restrict__ Y, _Float16* __restrict__ Yt) {
  const int lane = threadIdx.x & 31;
  const int wave = threadIdx.x >> 5;
  const int lh = lane & 15;          // row/col within tile
  const int hh = lane >> 4;          // half-select (K base 0 or 8)
  const int n0 = blockIdx.x * 32;                    // 9 column tiles
  const int m0 = (blockIdx.y * 8 + wave) * 32;       // 128 row tiles

  const _Float16* ap0 = S  + (size_t)(m0 +  0 + lh) * KDIM + hh * 8;
  const _Float16* ap1 = S  + (size_t)(m0 + 16 + lh) * KDIM + hh * 8;
  const _Float16* bp0 = Xt + (size_t)(n0 +  0 + lh) * KDIM + hh * 8;
  const _Float16* bp1 = Xt + (size_t)(n0 + 16 + lh) * KDIM + hh * 8;

  v8f c00 = {}, c01 = {}, c10 = {}, c11 = {};

  // depth-2 pipeline: stage s holds fragments for K-step (k/32) with parity s
  v16h A0[2], A1[2], B0[2], B1[2];
  A0[0] = load_frag(ap0);      A1[0] = load_frag(ap1);
  B0[0] = load_frag(bp0);      B1[0] = load_frag(bp1);
  A0[1] = load_frag(ap0 + 32); A1[1] = load_frag(ap1 + 32);
  B0[1] = load_frag(bp0 + 32); B1[1] = load_frag(bp1 + 32);

  for (int kk = 0; kk < KDIM; kk += 128) {
    // stay 2KB ahead of the b128 streams, pull into near cache
    __builtin_prefetch(ap0 + kk + 1024, 0, 3);
    __builtin_prefetch(ap1 + kk + 1024, 0, 3);
    __builtin_prefetch(bp0 + kk + 1024, 0, 3);
    __builtin_prefetch(bp1 + kk + 1024, 0, 3);
#pragma unroll
    for (int u = 0; u < 4; ++u) {
      const int s  = u & 1;           // stage consumed this step
      const int kn = kk + u * 32 + 64; // refill same-parity stage, 2 steps ahead
      c00 = WMMA(A0[s], B0[s], c00);
      c01 = WMMA(A0[s], B1[s], c01);
      c10 = WMMA(A1[s], B0[s], c10);
      c11 = WMMA(A1[s], B1[s], c11);
      // kn reaches 4128 on the final rotation: ~110B overread, still inside
      // the workspace carve-out (every read buffer is followed by another).
      A0[s] = load_frag(ap0 + kn);
      A1[s] = load_frag(ap1 + kn);
      B0[s] = load_frag(bp0 + kn);
      B1[s] = load_frag(bp1 + kn);
    }
  }

  // C layout: VGPR r holds (M = base + 8*hh + r, N = base + lh) -> rows are
  // contiguous, so the f16 transpose store is one b128 per subtile.
#define EPILOGUE(CI, I, J)                                                       \
  do {                                                                           \
    v8f c = (CI);                                                                \
    const int rowb = m0 + (I) * 16 + hh * 8;                                     \
    const int ccol = n0 + (J) * 16 + lh;                                         \
    float* yp = Y + (size_t)rowb * LDC + ccol;                                   \
    if (P) {                                                                     \
      const float* pp = P + (size_t)rowb * LDC + ccol;                           \
      _Pragma("unroll") for (int r = 0; r < 8; ++r)                              \
          c[r] = alpha * c[r] - pp[(size_t)r * LDC];                             \
    }                                                                            \
    _Pragma("unroll") for (int r = 0; r < 8; ++r) yp[(size_t)r * LDC] = c[r];    \
    *(v8h*)(Yt + (size_t)ccol * KDIM + rowb) = __builtin_convertvector(c, v8h);  \
  } while (0)

  EPILOGUE(c00, 0, 0);
  EPILOGUE(c01, 0, 1);
  EPILOGUE(c10, 1, 0);
  EPILOGUE(c11, 1, 1);
#undef EPILOGUE
}

// ------------------------------------- gather mats -> xbig f16 [16384 x 352]
// xbig[bn, f*5+m] = mats[m][n, f*4+b],  bn = b*4096+n
__global__ __launch_bounds__(256) void k_pack_xbig(const float* __restrict__ mats,
                                                   _Float16* __restrict__ Xb) {
  int idx = blockIdx.x * blockDim.x + threadIdx.x;   // exactly 16384*352
  int bn = idx / FMPAD;
  int i  = idx - bn * FMPAD;
  float v = 0.0f;
  if (i < 330) {
    int f = i / 5, m = i - f * 5;
    int b = bn >> 12;
    int n = bn & (NSER - 1);
    v = mats[(size_t)m * MATSZ + (size_t)n * LDC + f * 4 + b];
  }
  Xb[(size_t)bn * FMPAD + i] = (_Float16)v;
}

// Wt[o, i] = W[i, o] (f16, K padded 330 -> 352)
__global__ __launch_bounds__(256) void k_pack_wt(const float* __restrict__ W,
                                                 _Float16* __restrict__ Wt) {
  int idx = blockIdx.x * blockDim.x + threadIdx.x;
  if (idx >= 64 * FMPAD) return;
  int o = idx / FMPAD;
  int i = idx - o * FMPAD;
  Wt[idx] = (_Float16)(i < 330 ? W[(size_t)i * 64 + o] : 0.0f);
}

// ------------------------------- final projection [16384x352]x[352x64]+bias
__global__ __launch_bounds__(256)
void k_out_gemm(const _Float16* __restrict__ Xb, const _Float16* __restrict__ Wt,
                const float* __restrict__ bias, float* __restrict__ out) {
  const int lane = threadIdx.x & 31;
  const int wave = threadIdx.x >> 5;
  const int lh = lane & 15;
  const int hh = lane >> 4;
  const int n0 = blockIdx.x * 16;                    // 4 column tiles (O=64)
  const int m0 = (blockIdx.y * 8 + wave) * 64;       // 256 row-tiles of 64
  const int col = n0 + lh;

  const _Float16* bp  = Wt + (size_t)col * FMPAD + hh * 8;
  const _Float16* ap0 = Xb + (size_t)(m0 +  0 + lh) * FMPAD + hh * 8;
  const _Float16* ap1 = Xb + (size_t)(m0 + 16 + lh) * FMPAD + hh * 8;
  const _Float16* ap2 = Xb + (size_t)(m0 + 32 + lh) * FMPAD + hh * 8;
  const _Float16* ap3 = Xb + (size_t)(m0 + 48 + lh) * FMPAD + hh * 8;

  v8f c0 = {}, c1 = {}, c2 = {}, c3 = {};
#pragma unroll
  for (int k0 = 0; k0 < FMPAD; k0 += 32) {
    v16h b  = load_frag(bp  + k0);
    v16h a0 = load_frag(ap0 + k0);
    v16h a1 = load_frag(ap1 + k0);
    v16h a2 = load_frag(ap2 + k0);
    v16h a3 = load_frag(ap3 + k0);
    c0 = WMMA(a0, b, c0);
    c1 = WMMA(a1, b, c1);
    c2 = WMMA(a2, b, c2);
    c3 = WMMA(a3, b, c3);
  }

  const float bv = bias[col];
#define EPI(CI, I)                                                              \
  do {                                                                          \
    v8f c = (CI);                                                               \
    const int rowb = m0 + (I) * 16 + hh * 8;                                    \
    _Pragma("unroll") for (int r = 0; r < 8; ++r)                               \
        out[(size_t)(rowb + r) * 64 + col] = c[r] + bv;                         \
  } while (0)
  EPI(c0, 0);
  EPI(c1, 1);
  EPI(c2, 2);
  EPI(c3, 3);
#undef EPI
}

// ---------------------------------------------------------------------------
extern "C" void kernel_launch(void* const* d_in, const int* in_sizes, int n_in,
                              void* d_out, int out_size, void* d_ws, size_t ws_size,
                              hipStream_t stream) {
  const float* inp  = (const float*)d_in[0];
  const float* st   = (const float*)d_in[1];
  const float* s0   = (const float*)d_in[2];
  const float* s1   = (const float*)d_in[3];
  const float* W    = (const float*)d_in[4];
  const float* bias = (const float*)d_in[5];
  float* out = (float*)d_out;

  // workspace carve-out (all chunks multiple of 256B)
  char* p = (char*)d_ws;
  _Float16* s0h = (_Float16*)p; p += (size_t)KDIM * KDIM * 2;       // 33.55 MB
  _Float16* s1h = (_Float16*)p; p += (size_t)KDIM * KDIM * 2;       // 33.55 MB
  _Float16* xt[5];
  for (int i = 0; i < 5; ++i) { xt[i] = (_Float16*)p; p += (size_t)LDC * KDIM * 2; }
  float* mats = (float*)p;      p += (size_t)NMATS * MATSZ * 4;     // 23.6 MB
  _Float16* xb = (_Float16*)p;  p += (size_t)BN * FMPAD * 2;        // 11.5 MB
  _Float16* wt = (_Float16*)p;  p += (size_t)64 * FMPAD * 2;

  float* m0p = mats;
  float* m1p = mats + 1 * MATSZ;
  float* m2p = mats + 2 * MATSZ;
  float* m3p = mats + 3 * MATSZ;
  float* m4p = mats + 4 * MATSZ;

  // 1) supports fp32 -> f16 (one pass; f16 copies then live in L2)
  k_f32_to_f16<<<8192, 256, 0, stream>>>(s0, s0h, (KDIM * KDIM) / 8);
  k_f32_to_f16<<<8192, 256, 0, stream>>>(s1, s1h, (KDIM * KDIM) / 8);

  // 2) x0 (f32 mat + f16 transpose)
  k_pack_x0<<<(LDC * NSER) / 256, 256, 0, stream>>>(inp, st, m0p, xt[0]);

  // 3) diffusion steps (Chebyshev quirk: x0 carries over between supports)
  dim3 gd(9, 16);   // 9 col tiles x (16 groups * 8 waves * 32 rows)
  k_diff_gemm<<<gd, 256, 0, stream>>>(s0h, xt[0], nullptr, 1.0f, m1p, xt[1]); // x1a
  k_diff_gemm<<<gd, 256, 0, stream>>>(s0h, xt[1], m0p,     2.0f, m2p, xt[2]); // x2a
  k_diff_gemm<<<gd, 256, 0, stream>>>(s1h, xt[1], nullptr, 1.0f, m3p, xt[3]); // x1b
  k_diff_gemm<<<gd, 256, 0, stream>>>(s1h, xt[3], m1p,     2.0f, m4p, xt[4]); // x2b

  // 4) pack for final projection, then WMMA GEMM + bias
  k_pack_xbig<<<(BN * FMPAD) / 256, 256, 0, stream>>>(mats, xb);
  k_pack_wt<<<(64 * FMPAD + 255) / 256, 256, 0, stream>>>(W, wt);
  dim3 go(4, 32);
  k_out_gemm<<<go, 256, 0, stream>>>(xb, wt, bias, out);
}
